// S4Model_70652212019538
// MI455X (gfx1250) — matrix operand order repaired and checked by reading, not compiled, not run on hardware
//
#include <hip/hip_runtime.h>
#include <hip/hip_bf16.h>
#include <math.h>

// ---------------- model dims ----------------
#define Hd     256
#define NHd    32
#define NLd    4
#define DOUTd  10
#define Bd     32
#define Ld     4096
#define TWOH   512

typedef __attribute__((ext_vector_type(16))) _Float16 v16h;
typedef __attribute__((ext_vector_type(8)))  _Float16 v8h;
typedef __attribute__((ext_vector_type(8)))  float    v8f;
typedef __attribute__((ext_vector_type(4)))  float    v4f;

static __device__ inline v16h cat8(v8h a, v8h b) {
  return __builtin_shufflevector(a, b, 0,1,2,3,4,5,6,7,8,9,10,11,12,13,14,15);
}

// ---------------------------------------------------------------------------
// Prep 1: discretize S4D poles.  w = exp(dt*A),  Ct = 2*C*expm1(dt*A)/A
// expm1(dtA) = w - 1 (complex).  One thread per (layer, h, n).
// ---------------------------------------------------------------------------
__global__ void __launch_bounds__(256) prep_params(
    const float* __restrict__ log_dt, const float* __restrict__ C_re,
    const float* __restrict__ C_im,   const float* __restrict__ log_A_real,
    const float* __restrict__ A_imag,
    float* __restrict__ wre, float* __restrict__ wim,
    float* __restrict__ ctre, float* __restrict__ ctim) {
  int idx = blockIdx.x * blockDim.x + threadIdx.x;        // NL*H*NH
  int i  = idx / (Hd * NHd);
  int hh = (idx / NHd) % Hd;
  float dt  = expf(log_dt[i * Hd + hh]);
  float Are = -expf(log_A_real[idx]);
  float Aim = A_imag[idx];
  float dre = Are * dt, dim = Aim * dt;
  float er  = expf(dre);
  float sn, cs; sincosf(dim, &sn, &cs);
  float w_r = er * cs, w_i = er * sn;                     // exp(dtA)
  float em_r = w_r - 1.0f, em_i = w_i;                    // expm1(dtA)
  float cr = C_re[idx], ci = C_im[idx];
  float nre = cr * em_r - ci * em_i;                      // C * expm1
  float nim = cr * em_i + ci * em_r;
  float den = 2.0f / (Are * Are + Aim * Aim);             // fold the 2x here
  ctre[idx] = (nre * Are + nim * Aim) * den;              // (C*expm1)/A
  ctim[idx] = (nim * Are - nre * Aim) * den;
  wre[idx] = w_r; wim[idx] = w_i;
}

// ---------------------------------------------------------------------------
// Prep 2: split out_w into f16 hi/lo pair (f16x3 error-compensated WMMA).
// ---------------------------------------------------------------------------
__global__ void __launch_bounds__(256) prep_w_split(
    const float* __restrict__ out_w, _Float16* __restrict__ owhi,
    _Float16* __restrict__ owlo) {
  int idx = blockIdx.x * blockDim.x + threadIdx.x;        // NL*2H*H
  float v = out_w[idx];
  _Float16 hi = (_Float16)v;
  owhi[idx] = hi;
  owlo[idx] = (_Float16)(v - (float)hi);
}

// ---------------------------------------------------------------------------
// Encoder: h[b,hh,l] = x[b,l]*enc_w[hh] + enc_b[hh]   (layout: l contiguous)
// ---------------------------------------------------------------------------
__global__ void __launch_bounds__(256) encoder_kernel(
    const float* __restrict__ x, const float* __restrict__ enc_w,
    const float* __restrict__ enc_b, float* __restrict__ hbuf) {
  size_t idx = (size_t)blockIdx.x * blockDim.x + threadIdx.x;   // B*H*L
  int l  = (int)(idx % Ld);
  int bh = (int)(idx / Ld);
  int hh = bh % Hd;
  int b  = bh / Hd;
  hbuf[idx] = x[(size_t)b * Ld + l] * enc_w[hh] + enc_b[hh];
}

// ---------------------------------------------------------------------------
// S4D scan: one wave per (b,h) channel, lane n owns complex state s_n.
// Per 32-step chunk:
//   - u[l] broadcast by v_readlane (compile-time lane index, no DS traffic)
//   - lane n keeps Re(Ct_n * s_n[l]) in a 32-entry register array
//   - one 32x32 LDS transpose per chunk (scattered b32 stores -> conflict-free,
//     contiguous b128 row loads), then 8 packed v4 adds give y for 32 l's
// y = gelu(conv + D*u) stored as f16 hi/lo split for the WMMA GEMM.
// ---------------------------------------------------------------------------
__global__ void __launch_bounds__(256) s4d_scan(
    const float* __restrict__ hbuf,
    const float* __restrict__ wre, const float* __restrict__ wim,
    const float* __restrict__ ctre, const float* __restrict__ ctim,
    const float* __restrict__ Dv,
    _Float16* __restrict__ yhi, _Float16* __restrict__ ylo, int layer) {
  __shared__ float Pt[8][32][36];   // per-wave 32x32 transpose pad->36 (16B rows)

  int wv   = threadIdx.x >> 5;
  int lane = threadIdx.x & 31;
  int wid  = blockIdx.x * 8 + wv;                  // 0..B*H-1
  int b  = wid / Hd;
  int hh = wid % Hd;
  int pb = layer * Hd * NHd + hh * NHd + lane;
  float w_r = wre[pb], w_i = wim[pb];
  float c_r = ctre[pb], c_in = -ctim[pb];
  float Dh = Dv[layer * Hd + hh];
  size_t rbase = (size_t)(b * Hd + hh) * Ld;
  float sre = 0.0f, sim = 0.0f;

  #pragma unroll 1
  for (int l0 = 0; l0 < Ld; l0 += 32) {
    float uvec = hbuf[rbase + l0 + lane];          // 32 inputs staged across lanes
    float p[32];
    #pragma unroll
    for (int j = 0; j < 32; ++j) {
      float ul = __uint_as_float(
          __builtin_amdgcn_readlane(__float_as_uint(uvec), j));  // u[l0+j], uniform
      float t   = fmaf(-w_i, sim, ul);
      float nre = fmaf(w_r, sre, t);               // s = w*s + u
      float nim = fmaf(w_r, sim, w_i * sre);
      sre = nre; sim = nim;
      p[j] = fmaf(c_in, sim, c_r * sre);           // Re(Ct * s)
    }
    // ---- transpose P[n][j] -> column-major in LDS, rows become l-index ----
    #pragma unroll
    for (int j = 0; j < 32; ++j)
      Pt[wv][j][lane] = p[j];                      // lane-consecutive: no conflicts
    const v4f* prow = (const v4f*)&Pt[wv][lane][0];
    v4f a4 = prow[0];
    #pragma unroll
    for (int q = 1; q < 8; ++q) a4 += prow[q];
    float yconv = (a4[0] + a4[1]) + (a4[2] + a4[3]);

    float ypre = fmaf(Dh, uvec, yconv);
    float g = 0.5f * ypre * (1.0f + erff(ypre * 0.70710678118654752f));
    _Float16 hi = (_Float16)g;
    yhi[rbase + l0 + lane] = hi;
    ylo[rbase + l0 + lane] = (_Float16)(g - (float)hi);
  }
}

// ---------------------------------------------------------------------------
// Fused: v = out_w @ y + out_b  (WMMA f16x3, f32 acc) -> GLU -> +residual
//        -> LayerNorm over H -> h (in place).
// Grid: (L/16, B).  Block: 256 = 8 waves.  Wave w owns rows [64w,64w+64),
// full K=256, N=16 columns (one l-tile).  Waves 4..7 produce the sigmoid
// gate rows (o>=256) and pass them to waves 0..3 through LDS.
// ---------------------------------------------------------------------------
__global__ void __launch_bounds__(256) s4d_out_glu_ln(
    const _Float16* __restrict__ owhi, const _Float16* __restrict__ owlo,
    const _Float16* __restrict__ yhi,  const _Float16* __restrict__ ylo,
    const float* __restrict__ out_b,
    const float* __restrict__ ln_g, const float* __restrict__ ln_b,
    float* __restrict__ hbuf, int layer) {
  __shared__ float sigS[Hd][17];   // sigmoid(gate) rows, padded vs bank conflicts
  __shared__ float zbS[Hd][17];    // z + residual, pre-LN
  __shared__ float red[16][17];    // tree-reduction scratch
  __shared__ float muS[16], rsS[16];

  int b   = blockIdx.y;
  int l0  = blockIdx.x * 16;
  int tid = threadIdx.x;
  int w    = tid >> 5;
  int lane = tid & 31;
  int m    = lane & 15;            // A row-in-tile / D column index n
  int half = lane >> 4;
  int o0   = w * 64;

  const _Float16* owhiL = owhi + (size_t)layer * TWOH * Hd;
  const _Float16* owloL = owlo + (size_t)layer * TWOH * Hd;
  size_t ybase = (size_t)b * Hd * Ld + l0;

  v8f acc[4] = {};
  #pragma unroll 1
  for (int k0 = 0; k0 < Hd; k0 += 32) {
    // ---- B fragment: lane = column l (m), 16 k values strided by Ld ----
    const _Float16* brh = yhi + ybase + (size_t)k0 * Ld + m;
    const _Float16* brl = ylo + ybase + (size_t)k0 * Ld + m;
    if (k0 + 32 < Hd) {
      __builtin_prefetch(yhi + ybase + (size_t)(k0 + 32 + lane) * Ld, 0, 3);
      __builtin_prefetch(ylo + ybase + (size_t)(k0 + 32 + lane) * Ld, 0, 3);
    }
    v16h bh, bl;
    #pragma unroll
    for (int s = 0; s < 16; ++s) {
      int kk = (s < 8) ? (8 * half + s) : (16 + 8 * half + (s - 8));
      bh[s] = brh[(size_t)kk * Ld];
      bl[s] = brl[(size_t)kk * Ld];
    }
    #pragma unroll
    for (int t = 0; t < 4; ++t) {
      // ---- A fragment: row o, two contiguous 8-wide k runs ----
      int o = o0 + t * 16 + m;
      const _Float16* arh = owhiL + (size_t)o * Hd + k0;
      const _Float16* arl = owloL + (size_t)o * Hd + k0;
      v16h ah = cat8(*(const v8h*)(arh + 8 * half),
                     *(const v8h*)(arh + 16 + 8 * half));
      v16h al = cat8(*(const v8h*)(arl + 8 * half),
                     *(const v8h*)(arl + 16 + 8 * half));
      // f16x3: Alo*Bhi + Ahi*Blo + Ahi*Bhi, f32 accumulate
      acc[t] = __builtin_amdgcn_wmma_f32_16x16x32_f16(
          false, al, false, bh, (short)0, acc[t], false, false);
      acc[t] = __builtin_amdgcn_wmma_f32_16x16x32_f16(
          false, ah, false, bl, (short)0, acc[t], false, false);
      acc[t] = __builtin_amdgcn_wmma_f32_16x16x32_f16(
          false, ah, false, bh, (short)0, acc[t], false, false);
    }
  }

  // ---- epilogue: bias + GLU + residual + LayerNorm ----
  const float* obL = out_b + layer * TWOH;
  if (w >= 4) {                    // gate half (o in [256,512))
    #pragma unroll
    for (int t = 0; t < 4; ++t)
      #pragma unroll
      for (int r = 0; r < 8; ++r) {
        int o = o0 + t * 16 + r + 8 * half;
        float v = acc[t][r] + obL[o];
        sigS[o - Hd][m] = 1.0f / (1.0f + expf(-v));
      }
  }
  __syncthreads();
  if (w < 4) {                     // value half (o in [0,256))
    #pragma unroll
    for (int t = 0; t < 4; ++t)
      #pragma unroll
      for (int r = 0; r < 8; ++r) {
        int o = o0 + t * 16 + r + 8 * half;
        float a = acc[t][r] + obL[o];
        float z = a * sigS[o][m];
        zbS[o][m] = z + hbuf[(size_t)(b * Hd + o) * Ld + l0 + m];
      }
  }
  __syncthreads();

  // ---- LayerNorm stats: 16 groups x 16 columns tree reduction ----
  int c = tid & 15, ggrp = tid >> 4;
  {
    float s = 0.0f;
    #pragma unroll
    for (int r = 0; r < 16; ++r) s += zbS[ggrp * 16 + r][c];
    red[ggrp][c] = s;
  }
  __syncthreads();
  if (tid < 16) {
    float s = 0.0f;
    #pragma unroll
    for (int g2 = 0; g2 < 16; ++g2) s += red[g2][tid];
    muS[tid] = s * (1.0f / Hd);
  }
  __syncthreads();
  {
    float mu = muS[c];
    float v = 0.0f;
    #pragma unroll
    for (int r = 0; r < 16; ++r) {
      float d = zbS[ggrp * 16 + r][c] - mu;
      v = fmaf(d, d, v);
    }
    red[ggrp][c] = v;
  }
  __syncthreads();
  if (tid < 16) {
    float v = 0.0f;
    #pragma unroll
    for (int g2 = 0; g2 < 16; ++g2) v += red[g2][tid];
    rsS[tid] = rsqrtf(v * (1.0f / Hd) + 1e-5f);
  }
  __syncthreads();
  {                                // normalize + affine, write h back
    int o = tid;
    float g  = ln_g[layer * Hd + o];
    float be = ln_b[layer * Hd + o];
    float* hrow = hbuf + (size_t)(b * Hd + o) * Ld + l0;
    #pragma unroll
    for (int cc = 0; cc < 16; ++cc)
      hrow[cc] = (zbS[o][cc] - muS[cc]) * rsS[cc] * g + be;
  }
}

// ---------------------------------------------------------------------------
// Mean-pool over L: one wave per (b,h) row.
// ---------------------------------------------------------------------------
__global__ void __launch_bounds__(256) pool_kernel(
    const float* __restrict__ hbuf, float* __restrict__ pooled) {
  int wid  = blockIdx.x * (blockDim.x >> 5) + (threadIdx.x >> 5);
  int lane = threadIdx.x & 31;
  const float* row = hbuf + (size_t)wid * Ld;
  float s = 0.0f;
  for (int l = lane; l < Ld; l += 32) s += row[l];
  s += __shfl_xor(s, 16, 32);
  s += __shfl_xor(s,  8, 32);
  s += __shfl_xor(s,  4, 32);
  s += __shfl_xor(s,  2, 32);
  s += __shfl_xor(s,  1, 32);
  if (lane == 0) pooled[wid] = s * (1.0f / Ld);
}

// ---------------------------------------------------------------------------
// Decoder: (B,H) @ dec_w.T + dec_b -> (B,10).  Tiny.
// ---------------------------------------------------------------------------
__global__ void __launch_bounds__(256) decoder_kernel(
    const float* __restrict__ pooled, const float* __restrict__ dec_w,
    const float* __restrict__ dec_b, float* __restrict__ out) {
  int t = blockIdx.x * blockDim.x + threadIdx.x;
  if (t >= Bd * DOUTd) return;
  int b = t / DOUTd, o = t % DOUTd;
  float s = dec_b[o];
  for (int k = 0; k < Hd; ++k)
    s = fmaf(pooled[b * Hd + k], dec_w[o * Hd + k], s);
  out[t] = s;
}

// ---------------------------------------------------------------------------
extern "C" void kernel_launch(void* const* d_in, const int* in_sizes, int n_in,
                              void* d_out, int out_size, void* d_ws, size_t ws_size,
                              hipStream_t stream) {
  (void)in_sizes; (void)n_in; (void)out_size; (void)ws_size;
  const float* x          = (const float*)d_in[0];
  const float* enc_w      = (const float*)d_in[1];
  const float* enc_b      = (const float*)d_in[2];
  const float* log_dt     = (const float*)d_in[3];
  const float* C_re       = (const float*)d_in[4];
  const float* C_im       = (const float*)d_in[5];
  const float* log_A_real = (const float*)d_in[6];
  const float* A_imag     = (const float*)d_in[7];
  const float* Dv         = (const float*)d_in[8];
  const float* out_w      = (const float*)d_in[9];
  const float* out_b      = (const float*)d_in[10];
  const float* ln_g       = (const float*)d_in[11];
  const float* ln_b       = (const float*)d_in[12];
  const float* dec_w      = (const float*)d_in[13];
  const float* dec_b      = (const float*)d_in[14];

  // ---- workspace layout (~271 MB) ----
  const size_t SZ_H   = (size_t)Bd * Hd * Ld * sizeof(float);      // 128 MB
  const size_t SZ_Y16 = (size_t)Bd * Hd * Ld * sizeof(_Float16);   //  64 MB
  const size_t SZ_P   = (size_t)NLd * Hd * NHd * sizeof(float);    // 128 KB
  const size_t SZ_OW  = (size_t)NLd * TWOH * Hd * sizeof(_Float16);//   1 MB
  char* ws = (char*)d_ws;
  float*    hbuf  = (float*)ws;                       ws += SZ_H;
  _Float16* yhi   = (_Float16*)ws;                    ws += SZ_Y16;
  _Float16* ylo   = (_Float16*)ws;                    ws += SZ_Y16;
  float*    wre   = (float*)ws;                       ws += SZ_P;
  float*    wim   = (float*)ws;                       ws += SZ_P;
  float*    ctre  = (float*)ws;                       ws += SZ_P;
  float*    ctim  = (float*)ws;                       ws += SZ_P;
  _Float16* owhi  = (_Float16*)ws;                    ws += SZ_OW;
  _Float16* owlo  = (_Float16*)ws;                    ws += SZ_OW;
  float*    pooled= (float*)ws;                       ws += (size_t)Bd * Hd * sizeof(float);

  prep_params<<<(NLd * Hd * NHd) / 256, 256, 0, stream>>>(
      log_dt, C_re, C_im, log_A_real, A_imag, wre, wim, ctre, ctim);
  prep_w_split<<<(NLd * TWOH * Hd) / 256, 256, 0, stream>>>(out_w, owhi, owlo);
  encoder_kernel<<<(Bd * Hd * (size_t)Ld) / 256, 256, 0, stream>>>(
      x, enc_w, enc_b, hbuf);

  for (int i = 0; i < NLd; ++i) {
    s4d_scan<<<(Bd * Hd) / 8, 256, 0, stream>>>(
        hbuf, wre, wim, ctre, ctim, Dv, yhi, ylo, i);
    s4d_out_glu_ln<<<dim3(Ld / 16, Bd), 256, 0, stream>>>(
        owhi, owlo, yhi, ylo, out_b, ln_g, ln_b, hbuf, i);
  }

  pool_kernel<<<(Bd * Hd) / 8, 256, 0, stream>>>(hbuf, pooled);
  decoder_kernel<<<2, 256, 0, stream>>>(pooled, dec_w, dec_b, (float*)d_out);
}
